// TransformerEncoder_48902497632264
// MI455X (gfx1250) — matrix operand order repaired and checked by reading, not compiled, not run on hardware
//
#include <hip/hip_runtime.h>
#include <hip/hip_bf16.h>
#include <stdint.h>

// ---------------------------------------------------------------------------
// Transformer encoder (L=4, D=1024, H=16, DH=64, FF=4096, tokens=2*2048=4096)
// bf16 WMMA (v_wmma_f32_16x16x32_bf16), fp32 accumulate.
// CDNA5 paths:
//   * GEMM tiles staged by the Tensor Data Mover (tensor_load_to_lds,
//     TENSORcnt) with LDS row padding done by the TDM descriptor.
//   * Attention tiles staged with global_load_async_to_lds_b128 (ASYNCcnt).
//   * Double-buffered GEMM mainloop, pre-transposed bf16 weights.
// ---------------------------------------------------------------------------

typedef __attribute__((ext_vector_type(16))) __bf16   bf16x16;
typedef __attribute__((ext_vector_type(8)))  float    f32x8;
typedef __attribute__((ext_vector_type(4)))  uint32_t u32x4;
typedef __attribute__((ext_vector_type(8)))  uint32_t u32x8;

#define LAYERS 4
#define DMODEL 1024
#define NHEAD  16
#define DHEAD  64
#define FFDIM  4096
#define SEQ    2048
#define NTOK   4096   // B*S
#define LNEPS  1e-5f

union FragBF16 {                 // 32 bytes = 16 bf16 = one WMMA A/B operand
    bf16x16 v;
    uint4   q[2];
    unsigned short u[16];
};

__device__ __forceinline__ unsigned short f2bf(float f) {
    union { float f; uint32_t u; } c; c.f = f;
    uint32_t r = c.u + 0x7FFFu + ((c.u >> 16) & 1u);   // RNE
    return (unsigned short)(r >> 16);
}

__device__ __forceinline__ unsigned ldsOffset(const void* p) {
    // addrspace(3) -> flat keeps the LDS byte offset in the low 32 bits
    return (unsigned)(unsigned long long)p;
}

__device__ __forceinline__ void asyncLoadB128(unsigned ldsAddr, unsigned vOff,
                                              unsigned long long sBase) {
    asm volatile("global_load_async_to_lds_b128 %0, %1, %2"
                 :: "v"(ldsAddr), "v"(vOff), "s"(sBase) : "memory");
}
__device__ __forceinline__ void waitAsync0() {
    asm volatile("s_wait_asynccnt 0x0" ::: "memory");
}

// ---------------------------------------------------------------------------
// TDM: load a 128-row x 32-bf16 tile (row stride K elements) from global into
// LDS, padding each 64B row with 16B (-> 80B LDS row stride == 40 halves).
// D# per CDNA5 ISA ch.8: group0 {count, lds_addr, global_addr, type=2},
// group1 {data_size=2B, pad_enable, pad_interval=16DW, pad_amount=4DW,
//         tensor_dim0=K, tensor_dim1=128, tile_dim0=32, tile_dim1=128,
//         tensor_dim0_stride=K}. Issued once per wave; tracked by TENSORcnt.
// ---------------------------------------------------------------------------
__device__ __forceinline__ void tdmLoadTile128x32(unsigned ldsAddr,
                                                  unsigned long long gAddr,
                                                  int K) {
    u32x4 g0;
    g0[0] = 1u;                                            // count=1, user mode
    g0[1] = ldsAddr;                                       // lds_addr (bytes)
    g0[2] = (uint32_t)gAddr;                               // global_addr[31:0]
    g0[3] = (uint32_t)((gAddr >> 32) & 0x01FFFFFFull)      // global_addr[56:32]
          | (2u << 30);                                    // type=2 ("image")
    u32x8 g1;
    g1[0] = (1u << 16)                                     // data_size=1 -> 2B
          | (1u << 20)                                     // pad_enable
          | (3u << 22)                                     // pad_interval: 16 DW
          | (3u << 25);                                    // pad_amount: 4 DW
    g1[1] = ((uint32_t)K & 0xFFFFu) << 16;                 // tensor_dim0 lo16
    g1[2] = ((uint32_t)K >> 16) | (128u << 16);            // dim0 hi / dim1 lo
    g1[3] = (32u << 16);                                   // tile_dim0 = 32
    g1[4] = 128u;                                          // tile_dim1 = 128
    g1[5] = (uint32_t)K;                                   // dim0_stride[31:0]
    g1[6] = 0u;
    g1[7] = 0u;
    asm volatile("tensor_load_to_lds %0, %1" :: "s"(g0), "s"(g1) : "memory");
}

// ---------------------------------------------------------------------------
// fp32 W[K][N] -> bf16 Bt[N][K]  (transpose + convert, 32x32 tiles via LDS)
// grid = (N/32, K/32), 256 threads
// ---------------------------------------------------------------------------
__global__ __launch_bounds__(256)
void cvt_tr_f32_bf16(const float* __restrict__ W, unsigned short* __restrict__ Bt,
                     int K, int N) {
    __shared__ unsigned short T[32][36];
    const int k0 = blockIdx.y * 32, n0 = blockIdx.x * 32;
    const int t  = threadIdx.x;
    {
        int r = t >> 3, cg = t & 7;                    // 32 rows x 8 col-groups
        float4 v = *(const float4*)(W + (size_t)(k0 + r) * N + n0 + cg * 4);
        T[r][cg * 4 + 0] = f2bf(v.x);
        T[r][cg * 4 + 1] = f2bf(v.y);
        T[r][cg * 4 + 2] = f2bf(v.z);
        T[r][cg * 4 + 3] = f2bf(v.w);
    }
    __syncthreads();
    {
        int nr = t >> 3, kg = t & 7;
        unsigned short o0 = T[kg * 4 + 0][nr], o1 = T[kg * 4 + 1][nr];
        unsigned short o2 = T[kg * 4 + 2][nr], o3 = T[kg * 4 + 3][nr];
        uint2 p;
        p.x = (uint32_t)o0 | ((uint32_t)o1 << 16);
        p.y = (uint32_t)o2 | ((uint32_t)o3 << 16);
        *(uint2*)(Bt + (size_t)(n0 + nr) * K + k0 + kg * 4) = p;
    }
}

// ---------------------------------------------------------------------------
// LayerNorm over D=1024, one block (256 threads) per token row.
// ---------------------------------------------------------------------------
template <int OUT_BF16>
__global__ __launch_bounds__(256)
void layernorm_k(const float* __restrict__ x, const float* __restrict__ wt,
                 const float* __restrict__ bs,
                 unsigned short* __restrict__ outH, float* __restrict__ outF) {
    const int row = blockIdx.x;
    const int tid = threadIdx.x;
    const float* xr = x + (size_t)row * DMODEL;
    float4 v = *(const float4*)(xr + tid * 4);
    float s1 = v.x + v.y + v.z + v.w;
    float s2 = v.x * v.x + v.y * v.y + v.z * v.z + v.w * v.w;
    #pragma unroll
    for (int off = 16; off > 0; off >>= 1) {
        s1 += __shfl_xor(s1, off);
        s2 += __shfl_xor(s2, off);
    }
    __shared__ float r1[8], r2[8];
    if ((tid & 31) == 0) { r1[tid >> 5] = s1; r2[tid >> 5] = s2; }
    __syncthreads();
    float t1 = 0.f, t2 = 0.f;
    #pragma unroll
    for (int i = 0; i < 8; i++) { t1 += r1[i]; t2 += r2[i]; }
    const float mean = t1 * (1.0f / DMODEL);
    const float var  = t2 * (1.0f / DMODEL) - mean * mean;
    const float inv  = rsqrtf(var + LNEPS);
    float e[4] = {v.x, v.y, v.z, v.w};
    #pragma unroll
    for (int k = 0; k < 4; k++) {
        int col = tid * 4 + k;
        float y = (e[k] - mean) * inv * wt[col] + bs[col];
        if (OUT_BF16) outH[(size_t)row * DMODEL + col] = f2bf(y);
        else          outF[(size_t)row * DMODEL + col] = y;
    }
}

// ---------------------------------------------------------------------------
// bf16 GEMM: C[M,N] = A[M,K] * Bt[N][K]^T + bias
//   MODE 0: fp32 out, += residual (fp32)
//   MODE 1: bf16 out
//   MODE 2: bf16 out, ReLU
//   MODE 3: bf16 out, stored transposed (out[n][m])  -- used for V projection
// Block tile 128x128, BLOCK_K=32, 256 threads = 8 waves (2x4), wave tile 64x32.
// Double-buffered LDS; tiles staged entirely by the TDM (wave0: A, wave1: B).
// ---------------------------------------------------------------------------
template <int MODE>
__global__ __launch_bounds__(256)
void gemm_bf16(const unsigned short* __restrict__ A,
               const unsigned short* __restrict__ B,   // Bt[N][K]
               const float* __restrict__ bias,
               const float* __restrict__ res,
               float* __restrict__ outF,
               unsigned short* __restrict__ outH,
               int M, int N, int K) {
    __shared__ unsigned short Alds[2][128 * 40];   // [row][k], padded stride 40
    __shared__ unsigned short Blds[2][128 * 40];   // Bt[n][k], padded stride 40

    const int tid  = threadIdx.x;
    const int lane = tid & 31;
    const int w    = tid >> 5;
    const int wm   = w & 1;          // 0..1  (64-row slab)
    const int wn   = w >> 1;         // 0..3  (32-col slab)
    const int l    = lane & 15;
    const int hi   = lane >> 4;
    const int mBase = blockIdx.y * 128;
    const int nBase = blockIdx.x * 128;

    const unsigned long long aPtr = (unsigned long long)(A + (size_t)mBase * K);
    const unsigned long long bPtr = (unsigned long long)(B + (size_t)nBase * K);

    f32x8 acc[4][2];
    #pragma unroll
    for (int i = 0; i < 4; i++)
        #pragma unroll
        for (int j = 0; j < 2; j++)
            #pragma unroll
            for (int e = 0; e < 8; e++) acc[i][j][e] = 0.f;

    auto issue = [&](int buf, int k0) {
        if (w == 0)
            tdmLoadTile128x32(ldsOffset(&Alds[buf][0]),
                              aPtr + (unsigned long long)k0 * 2, K);
        else if (w == 1)
            tdmLoadTile128x32(ldsOffset(&Blds[buf][0]),
                              bPtr + (unsigned long long)k0 * 2, K);
    };

    issue(0, 0);
    const int KT = K / 32;
    for (int kt = 0; kt < KT; kt++) {
        const int cur = kt & 1;
        if (w < 2) __builtin_amdgcn_s_wait_tensorcnt(0);
        __syncthreads();                 // tile `cur` resident, prev compute done
        if (kt + 1 < KT) issue(cur ^ 1, (kt + 1) * 32);

        FragBF16 af[4], bf[2];
        #pragma unroll
        for (int ms = 0; ms < 4; ms++) {
            int row = wm * 64 + ms * 16 + l;           // lane = row M
            af[ms].q[0] = *(const uint4*)(&Alds[cur][row * 40 + hi * 8]);
            af[ms].q[1] = *(const uint4*)(&Alds[cur][row * 40 + 16 + hi * 8]);
        }
        #pragma unroll
        for (int ns = 0; ns < 2; ns++) {
            int n = wn * 32 + ns * 16 + l;             // lane = col N
            bf[ns].q[0] = *(const uint4*)(&Blds[cur][n * 40 + hi * 8]);
            bf[ns].q[1] = *(const uint4*)(&Blds[cur][n * 40 + 16 + hi * 8]);
        }
        #pragma unroll
        for (int ms = 0; ms < 4; ms++)
            #pragma unroll
            for (int ns = 0; ns < 2; ns++)
                acc[ms][ns] = __builtin_amdgcn_wmma_f32_16x16x32_bf16(
                    false, af[ms].v, false, bf[ns].v, (short)0, acc[ms][ns], false, false);
    }

    // ---- epilogue ----
    #pragma unroll
    for (int ms = 0; ms < 4; ms++) {
        #pragma unroll
        for (int ns = 0; ns < 2; ns++) {
            int gn = nBase + wn * 32 + ns * 16 + l;
            float bv = bias[gn];
            #pragma unroll
            for (int r = 0; r < 8; r++) {
                int gm = mBase + wm * 64 + ms * 16 + hi * 8 + r;
                float v = acc[ms][ns][r] + bv;
                if (MODE == 0) {
                    size_t idx = (size_t)gm * N + gn;
                    outF[idx] = v + res[idx];
                } else if (MODE == 3) {
                    outH[(size_t)gn * M + gm] = f2bf(v);
                } else {
                    if (MODE == 2) v = fmaxf(v, 0.f);
                    outH[(size_t)gm * N + gn] = f2bf(v);
                }
            }
        }
    }
}

// ---------------------------------------------------------------------------
// Flash attention: block = (b*H + h, qTile of 64), 128 threads = 4 waves.
// K staged as-is; V comes in pre-transposed ([D][NTOK]) so both stage with
// async-to-LDS b128 and land directly in B-fragment layout.
// ---------------------------------------------------------------------------
__global__ __launch_bounds__(128)
void attn_k(const unsigned short* __restrict__ Q,
            const unsigned short* __restrict__ Km,
            const unsigned short* __restrict__ Vt,   // [DMODEL][NTOK]
            unsigned short* __restrict__ O) {
    __shared__ unsigned short Klds[64 * 72];   // [key][d]
    __shared__ unsigned short Vlds[64 * 72];   // [d][key]
    __shared__ unsigned short Plds[64 * 72];   // per-wave 16-row scratch

    const int bh    = blockIdx.x;
    const int b     = bh >> 4;
    const int h     = bh & 15;
    const int qBase = blockIdx.y * 64;
    const int tid   = threadIdx.x;
    const int lane  = tid & 31;
    const int w     = tid >> 5;                // wave 0..3
    const int l     = lane & 15;
    const int hi    = lane >> 4;
    const int colBase = h * DHEAD;
    const float scale = 0.125f;                // 1/sqrt(64)

    const unsigned kB = ldsOffset(&Klds[0]);
    const unsigned vB = ldsOffset(&Vlds[0]);
    const unsigned long long kPtr =
        (unsigned long long)(Km + (size_t)b * SEQ * DMODEL + colBase);
    const unsigned long long vPtr =
        (unsigned long long)(Vt + (size_t)colBase * NTOK + (size_t)b * SEQ);

    // Q fragments for this wave's 16 rows (A layout: lane = row)
    FragBF16 qf[2];
    {
        size_t r = (size_t)(b * SEQ + qBase + w * 16 + l) * DMODEL + colBase;
        #pragma unroll
        for (int kk = 0; kk < 2; kk++) {
            qf[kk].q[0] = *(const uint4*)(Q + r + kk * 32 + hi * 8);
            qf[kk].q[1] = *(const uint4*)(Q + r + kk * 32 + hi * 8 + 16);
        }
    }

    f32x8 accO[4];
    #pragma unroll
    for (int j = 0; j < 4; j++)
        #pragma unroll
        for (int e = 0; e < 8; e++) accO[j][e] = 0.f;
    float mst[8], lst[8];
    #pragma unroll
    for (int r = 0; r < 8; r++) { mst[r] = -3.0e38f; lst[r] = 0.f; }

    for (int kb = 0; kb < SEQ / 64; kb++) {
        __syncthreads();                 // previous iteration's reads done
        // stage K[key][d] and Vt[d][key] blocks: 4 x 16B chunks each per thread
        #pragma unroll
        for (int rep = 0; rep < 4; rep++) {
            int id  = rep * 128 + tid;
            int row = id >> 3, c = id & 7;
            unsigned kv = (unsigned)((((size_t)(kb * 64 + row)) * DMODEL + c * 8) * 2);
            asyncLoadB128(kB + (unsigned)(row * 144 + c * 16), kv, kPtr);
            unsigned vv = (unsigned)((((size_t)row) * NTOK + kb * 64 + c * 8) * 2);
            asyncLoadB128(vB + (unsigned)(row * 144 + c * 16), vv, vPtr);
        }
        waitAsync0();
        __syncthreads();

        // S = Q * K^T  (K rows are already B-fragment layout)
        f32x8 s[4];
        #pragma unroll
        for (int j = 0; j < 4; j++)
            #pragma unroll
            for (int e = 0; e < 8; e++) s[j][e] = 0.f;
        #pragma unroll
        for (int kk = 0; kk < 2; kk++) {
            #pragma unroll
            for (int j = 0; j < 4; j++) {
                FragBF16 bfr;
                int key = j * 16 + l;
                bfr.q[0] = *(const uint4*)(&Klds[key * 72 + kk * 32 + hi * 8]);
                bfr.q[1] = *(const uint4*)(&Klds[key * 72 + kk * 32 + hi * 8 + 16]);
                s[j] = __builtin_amdgcn_wmma_f32_16x16x32_bf16(
                    false, qf[kk].v, false, bfr.v, (short)0, s[j], false, false);
            }
        }

        // online softmax (row reductions inside 16-lane C-layout groups)
        float al[8];
        #pragma unroll
        for (int r = 0; r < 8; r++) {
            float m = -3.0e38f;
            #pragma unroll
            for (int j = 0; j < 4; j++) m = fmaxf(m, s[j][r]);
            #pragma unroll
            for (int off = 1; off < 16; off <<= 1) m = fmaxf(m, __shfl_xor(m, off));
            float mn = fmaxf(mst[r], m * scale);
            al[r] = __expf(mst[r] - mn);
            mst[r] = mn;
        }
        float rs[8];
        #pragma unroll
        for (int r = 0; r < 8; r++) rs[r] = 0.f;
        #pragma unroll
        for (int j = 0; j < 4; j++)
            #pragma unroll
            for (int r = 0; r < 8; r++) {
                float p = __expf(s[j][r] * scale - mst[r]);
                s[j][r] = p;
                rs[r] += p;
            }
        #pragma unroll
        for (int r = 0; r < 8; r++) {
            float t = rs[r];
            #pragma unroll
            for (int off = 1; off < 16; off <<= 1) t += __shfl_xor(t, off);
            lst[r] = lst[r] * al[r] + t;
            #pragma unroll
            for (int j = 0; j < 4; j++) accO[j][r] *= al[r];
        }

        // C-layout -> A-layout for P via per-wave LDS scratch
        #pragma unroll
        for (int j = 0; j < 4; j++)
            #pragma unroll
            for (int r = 0; r < 8; r++)
                Plds[(w * 16 + hi * 8 + r) * 72 + j * 16 + l] = f2bf(s[j][r]);
        asm volatile("s_wait_dscnt 0x0" ::: "memory");

        // O += P * V  (Vlds rows are the B-fragment layout)
        #pragma unroll
        for (int kk = 0; kk < 2; kk++) {
            FragBF16 pa;
            int prow = w * 16 + l;
            pa.q[0] = *(const uint4*)(&Plds[prow * 72 + kk * 32 + hi * 8]);
            pa.q[1] = *(const uint4*)(&Plds[prow * 72 + kk * 32 + hi * 8 + 16]);
            #pragma unroll
            for (int j = 0; j < 4; j++) {
                FragBF16 vbf;
                int d = j * 16 + l;
                vbf.q[0] = *(const uint4*)(&Vlds[d * 72 + kk * 32 + hi * 8]);
                vbf.q[1] = *(const uint4*)(&Vlds[d * 72 + kk * 32 + hi * 8 + 16]);
                accO[j] = __builtin_amdgcn_wmma_f32_16x16x32_bf16(
                    false, pa.v, false, vbf.v, (short)0, accO[j], false, false);
            }
        }
    }

    // write O (bf16), normalized by row sums
    #pragma unroll
    for (int j = 0; j < 4; j++) {
        #pragma unroll
        for (int r = 0; r < 8; r++) {
            int gr = b * SEQ + qBase + w * 16 + hi * 8 + r;
            int gc = colBase + j * 16 + l;
            O[(size_t)gr * DMODEL + gc] = f2bf(accO[j][r] / lst[r]);
        }
    }
}

// ---------------------------------------------------------------------------
// Host-side orchestration
// ---------------------------------------------------------------------------
extern "C" void kernel_launch(void* const* d_in, const int* in_sizes, int n_in,
                              void* d_out, int out_size, void* d_ws, size_t ws_size,
                              hipStream_t stream) {
    (void)in_sizes; (void)n_in; (void)out_size; (void)ws_size;
    const float* x    = (const float*)d_in[0];
    const float* Wq   = (const float*)d_in[1];
    const float* Wk   = (const float*)d_in[2];
    const float* Wv   = (const float*)d_in[3];
    const float* Wo   = (const float*)d_in[4];
    const float* bq   = (const float*)d_in[5];
    const float* bk   = (const float*)d_in[6];
    const float* bv   = (const float*)d_in[7];
    const float* bo   = (const float*)d_in[8];
    const float* W1   = (const float*)d_in[9];
    const float* b1   = (const float*)d_in[10];
    const float* W2   = (const float*)d_in[11];
    const float* b2   = (const float*)d_in[12];
    const float* ln1w = (const float*)d_in[13];
    const float* ln1b = (const float*)d_in[14];
    const float* ln2w = (const float*)d_in[15];
    const float* ln2b = (const float*)d_in[16];
    const float* lnfw = (const float*)d_in[17];
    const float* lnfb = (const float*)d_in[18];

    char* ws = (char*)d_ws;
    float*          xbuf = (float*)ws;                                   // 16 MB
    unsigned short* hbuf = (unsigned short*)(ws + (size_t)16 * 1048576); //  8 MB
    unsigned short* qbuf = (unsigned short*)(ws + (size_t)24 * 1048576); //  8 MB
    unsigned short* kbuf = (unsigned short*)(ws + (size_t)32 * 1048576); //  8 MB
    unsigned short* vbuf = (unsigned short*)(ws + (size_t)40 * 1048576); //  8 MB (V^T)
    unsigned short* obuf = (unsigned short*)(ws + (size_t)48 * 1048576); //  8 MB
    unsigned short* ubuf = (unsigned short*)(ws + (size_t)56 * 1048576); // 32 MB
    unsigned short* wbuf = (unsigned short*)(ws + (size_t)88 * 1048576); //  8 MB

    hipMemcpyAsync(xbuf, x, (size_t)NTOK * DMODEL * sizeof(float),
                   hipMemcpyDeviceToDevice, stream);

    const dim3 gD(DMODEL / 128, NTOK / 128);   // (8, 32)  D x D GEMMs
    const dim3 gF(FFDIM / 128, NTOK / 128);    // (32, 32) FFN1
    const dim3 gA(2 * NHEAD, SEQ / 64);        // (32, 32) attention
    const dim3 gTdd(DMODEL / 32, DMODEL / 32); // W[D][D]  -> Bt
    const dim3 gTdf(FFDIM / 32, DMODEL / 32);  // W1[D][FF] -> Bt
    const dim3 gTfd(DMODEL / 32, FFDIM / 32);  // W2[FF][D] -> Bt

    const int nDD = DMODEL * DMODEL;           // 1M
    const int nDF = DMODEL * FFDIM;            // 4M

    for (int i = 0; i < LAYERS; i++) {
        // ---- pre-norm attention ----
        layernorm_k<1><<<NTOK, 256, 0, stream>>>(xbuf, ln1w + i * DMODEL, ln1b + i * DMODEL,
                                                 hbuf, nullptr);
        cvt_tr_f32_bf16<<<gTdd, 256, 0, stream>>>(Wq + (size_t)i * nDD, wbuf, DMODEL, DMODEL);
        gemm_bf16<1><<<gD, 256, 0, stream>>>(hbuf, wbuf, bq + i * DMODEL, nullptr,
                                             nullptr, qbuf, NTOK, DMODEL, DMODEL);
        cvt_tr_f32_bf16<<<gTdd, 256, 0, stream>>>(Wk + (size_t)i * nDD, wbuf, DMODEL, DMODEL);
        gemm_bf16<1><<<gD, 256, 0, stream>>>(hbuf, wbuf, bk + i * DMODEL, nullptr,
                                             nullptr, kbuf, NTOK, DMODEL, DMODEL);
        cvt_tr_f32_bf16<<<gTdd, 256, 0, stream>>>(Wv + (size_t)i * nDD, wbuf, DMODEL, DMODEL);
        gemm_bf16<3><<<gD, 256, 0, stream>>>(hbuf, wbuf, bv + i * DMODEL, nullptr,
                                             nullptr, vbuf, NTOK, DMODEL, DMODEL);

        attn_k<<<gA, 128, 0, stream>>>(qbuf, kbuf, vbuf, obuf);

        cvt_tr_f32_bf16<<<gTdd, 256, 0, stream>>>(Wo + (size_t)i * nDD, wbuf, DMODEL, DMODEL);
        gemm_bf16<0><<<gD, 256, 0, stream>>>(obuf, wbuf, bo + i * DMODEL, xbuf,
                                             xbuf, nullptr, NTOK, DMODEL, DMODEL);

        // ---- pre-norm FFN ----
        layernorm_k<1><<<NTOK, 256, 0, stream>>>(xbuf, ln2w + i * DMODEL, ln2b + i * DMODEL,
                                                 hbuf, nullptr);
        cvt_tr_f32_bf16<<<gTdf, 256, 0, stream>>>(W1 + (size_t)i * nDF, wbuf, DMODEL, FFDIM);
        gemm_bf16<2><<<gF, 256, 0, stream>>>(hbuf, wbuf, b1 + i * FFDIM, nullptr,
                                             nullptr, ubuf, NTOK, FFDIM, DMODEL);
        cvt_tr_f32_bf16<<<gTfd, 256, 0, stream>>>(W2 + (size_t)i * nDF, wbuf, FFDIM, DMODEL);
        gemm_bf16<0><<<gD, 256, 0, stream>>>(ubuf, wbuf, b2 + i * DMODEL, xbuf,
                                             xbuf, nullptr, NTOK, DMODEL, FFDIM);
    }

    layernorm_k<0><<<NTOK, 256, 0, stream>>>(xbuf, lnfw, lnfb, nullptr, (float*)d_out);
}
